// FBNet_4466765988258
// MI455X (gfx1250) — matrix operand order repaired and checked by reading, not compiled, not run on hardware
//
#include <hip/hip_runtime.h>

// ---------------------------------------------------------------------------
// FBNet (frame interpolation) for gfx1250 / MI455X.
// Convs = implicit GEMM via v_wmma_f32_16x16x32_f16 (f16 inputs, f32 accum),
// 2 WMMAs per K-chunk of 64, weight tiles DMA'd to LDS with
// global_load_async_to_lds_b128 (ASYNCcnt) when the toolchain exposes it.
// Everything else = coalesced f32 elementwise kernels.
// ---------------------------------------------------------------------------

typedef __attribute__((ext_vector_type(16))) _Float16 v16h;
typedef __attribute__((ext_vector_type(8)))  _Float16 v8h;
typedef __attribute__((ext_vector_type(8)))  float    v8f;
typedef int v4i __attribute__((vector_size(16)));

union V16H { v16h v; v8h h[2]; };

#define ACT_LIN  0
#define ACT_RELU 1
#define ACT_CLIP 2

#define AS1 __attribute__((address_space(1)))
#define AS3 __attribute__((address_space(3)))

#if defined(__HIP_DEVICE_COMPILE__) && __has_builtin(__builtin_amdgcn_global_load_async_to_lds_b128)
#define FB_ASYNC 1
#else
#define FB_ASYNC 0
#endif

__device__ __forceinline__ void fb_async_wait() {
#if FB_ASYNC
#if __has_builtin(__builtin_amdgcn_s_wait_asynccnt)
    __builtin_amdgcn_s_wait_asynccnt(0);
#else
    asm volatile("s_wait_asynccnt 0" ::: "memory");
#endif
#endif
}

// ---------------------------------------------------------------------------
// Weight tiling: f32 HWIO weights -> f16 B-tiles [nt][kc][n=16][kk=64],
// zero-padded, ready for straight LDS copy. k = (ky*ks+kx)*Cin+ci.
// ---------------------------------------------------------------------------
__global__ void fbnet_wtile(const float* __restrict__ w, _Float16* __restrict__ dst,
                            int Ktot, int Cout, int nkc, int total) {
    int i = blockIdx.x * 256 + threadIdx.x;
    if (i >= total) return;
    int kk = i & 63;
    int t  = i >> 6;
    int n  = t & 15;
    t >>= 4;                      // t = nt*nkc + c
    int c  = t % nkc;
    int nt = t / nkc;
    int k  = c * 64 + kk;
    int co = nt * 16 + n;
    _Float16 v = (_Float16)0.f;
    if (k < Ktot && co < Cout) v = (_Float16)w[(size_t)k * Cout + co];
    dst[i] = v;
}

// ---------------------------------------------------------------------------
// WMMA implicit-GEMM convolution.
// in  : f32 NHWC dense [Bn,H,W,Cin]
// w16 : f16 tiled weights (fbnet_wtile layout)
// out : f32 NHWC [Bn,H,W,CoutTot], writes channels [Coff, Coff+Cout)
// Block = 128 threads = 4 waves; block tile = 64 pixels x 16 out-channels;
// each wave owns one 16x16 output tile; K consumed in chunks of 64 (2 WMMA).
// ---------------------------------------------------------------------------
__global__ __launch_bounds__(128)
void fbnet_conv_wmma(const float* __restrict__ in, const _Float16* __restrict__ w16,
                     const float* __restrict__ bias, float* __restrict__ out,
                     int Bn, int H_, int W_, int Cin, int Cout, int CoutTot, int Coff,
                     int ks, int pad, int act, int totalPix)
{
    __shared__ alignas(128) _Float16 Alds[4][16][64];   // [wave][M][K]
    __shared__ alignas(128) _Float16 Blds[16][64];      // [N][K] (B transposed)
    __shared__ int pixOff[64];                          // base f32 index of pixel
    __shared__ int pixYX[64];                           // (y<<16)|x, 0x80008000 = dead

    const int tid  = threadIdx.x;
    const int wave = tid >> 5;
    const int lane = tid & 31;
    const int mrow = lane & 15;     // M for A-frag, N for B-frag
    const int kh   = lane >> 4;     // k-half selector
    const int pixBase = blockIdx.x * 64;
    const int ntile   = blockIdx.y;
    const int Ktot    = ks * ks * Cin;
    const int nkc     = (Ktot + 63) >> 6;
    const int kk      = tid & 63;   // this thread's fixed K column

    // ---- per-block pixel decomposition table (once) ----
    if (tid < 64) {
        int pix = pixBase + tid;
        if (pix < totalPix) {
            int x  = pix % W_;
            int by = pix / W_;
            int y  = by % H_;
            int b  = by / H_;
            pixOff[tid] = ((b * H_ + y) * W_ + x) * Cin;
            pixYX[tid]  = (y << 16) | x;
        } else {
            pixOff[tid] = 0;
            pixYX[tid]  = (int)0x80008000;   // forces bounds check to fail
        }
    }

    v8f acc = {0.f, 0.f, 0.f, 0.f, 0.f, 0.f, 0.f, 0.f};
    __syncthreads();

    for (int c = 0; c < nkc; ++c) {
        const int kc = c << 6;
        const _Float16* bsrc = w16 + ((size_t)ntile * nkc + c) * 1024;

        // ---- B tile: contiguous 2KB weight tile -> LDS ----
#if FB_ASYNC
        if (wave == 0) {
#pragma unroll
            for (int i = 0; i < 4; ++i) {
                const char* gsrcb = (const char*)bsrc + i * 512 + lane * 16;
                char*       ldsb  = (char*)&Blds[0][0] + i * 512 + lane * 16;
                __builtin_amdgcn_global_load_async_to_lds_b128(
                    (AS1 v4i*)gsrcb, (AS3 v4i*)ldsb, 0, 0);
            }
        }
#else
        {
            const unsigned* s = (const unsigned*)bsrc;
            unsigned* d = (unsigned*)&Blds[0][0];
            for (int i = tid; i < 512; i += 128) d[i] = s[i];
        }
#endif
        // prefetch next weight tile (emits global_prefetch_b8)
        __builtin_prefetch(bsrc + 1024, 0, 0);

        // ---- per-thread k decomposition (one div/mod per chunk) ----
        int k = kc + kk;
        bool kval = k < Ktot;
        int ci  = k % Cin;
        int t   = k / Cin;
        int dxo = t % ks - pad;
        int dyo = t / ks - pad;
        int delta = (dyo * W_ + dxo) * Cin + ci;

        // ---- A tile (im2col, SAME padding); overlaps with weight DMA ----
        for (int m = (tid >> 6); m < 64; m += 2) {
            int yx = pixYX[m];
            int iy = (int)(short)(yx >> 16) + dyo;
            int ix = (int)(short)(yx & 0xFFFF) + dxo;
            float v = 0.f;
            if (kval && (unsigned)iy < (unsigned)H_ && (unsigned)ix < (unsigned)W_)
                v = in[pixOff[m] + delta];
            Alds[m >> 4][m & 15][kk] = (_Float16)v;
        }

#if FB_ASYNC
        if (wave == 0) fb_async_wait();
#endif
        __syncthreads();

        // ---- per-wave fragment loads (CDNA5 wave32 WMMA layouts) ----
        // A 16x32 f16: lane m = L%16; e0..7 -> k = kh*8 + 0..7; e8..15 -> 16 + kh*8 + 0..7
        V16H a0, a1, b0, b1;
        const _Float16* ap = &Alds[wave][mrow][0];
        a0.h[0] = *(const v8h*)(ap + kh * 8);
        a0.h[1] = *(const v8h*)(ap + 16 + kh * 8);
        a1.h[0] = *(const v8h*)(ap + 32 + kh * 8);
        a1.h[1] = *(const v8h*)(ap + 48 + kh * 8);
        // B 32x16 f16 stored [N][K]: lane n = L%16; e0..7 -> k = kh*8 + 0..7; e8..15 -> 16 + kh*8 + 0..7
        const _Float16* bp = &Blds[mrow][0];
        b0.h[0] = *(const v8h*)(bp + kh * 8);
        b0.h[1] = *(const v8h*)(bp + 16 + kh * 8);
        b1.h[0] = *(const v8h*)(bp + 32 + kh * 8);
        b1.h[1] = *(const v8h*)(bp + 48 + kh * 8);

        acc = __builtin_amdgcn_wmma_f32_16x16x32_f16(false, a0.v, false, b0.v,
                                                     (short)0, acc, false, false);
        acc = __builtin_amdgcn_wmma_f32_16x16x32_f16(false, a1.v, false, b1.v,
                                                     (short)0, acc, false, false);
        __syncthreads();
    }

    // ---- store: C/D layout: lanes0-15 VGPR r -> M=r; lanes16-31 -> M=8+r; N = lane%16
    int co = ntile * 16 + mrow;
    if (co < Cout) {
        float bv = bias ? bias[co] : 0.f;
#pragma unroll
        for (int r = 0; r < 8; ++r) {
            int m   = r + kh * 8;
            int pix = pixBase + wave * 16 + m;
            if (pix < totalPix) {
                float v = acc[r] + bv;
                if (act == ACT_RELU)      v = v > 0.f ? v : 0.f;
                else if (act == ACT_CLIP) v = fminf(fmaxf(v, 0.f), 1.f);
                out[(size_t)pix * CoutTot + Coff + co] = v;
            }
        }
    }
}

// ---------------------------------------------------------------------------
// Elementwise kernels
// ---------------------------------------------------------------------------
__global__ void fbnet_fill0(float* __restrict__ d, int n) {
    int i = blockIdx.x * 256 + threadIdx.x;
    if (i < n) d[i] = 0.f;
}

__global__ void fbnet_avgpool(const float* __restrict__ src, float* __restrict__ dst,
                              int Bn, int ho, int wo, int C,
                              int srcC, int srcOff, int dstC, int dstOff) {
    int i = blockIdx.x * 256 + threadIdx.x;
    int total = Bn * ho * wo * C;
    if (i >= total) return;
    int c = i % C; int t = i / C;
    int x = t % wo; t /= wo;
    int y = t % ho; int b = t / ho;
    int wi = wo * 2, hi = ho * 2;
    size_t s0 = ((size_t)(b * hi + 2 * y) * wi + 2 * x) * srcC + srcOff + c;
    size_t rs = (size_t)wi * srcC;
    float v = 0.25f * (src[s0] + src[s0 + srcC] + src[s0 + rs] + src[s0 + rs + srcC]);
    dst[((size_t)(b * ho + y) * wo + x) * dstC + dstOff + c] = v;
}

// bilinear x2 upsample, half-pixel centers, edge clamp (jax.image.resize 'bilinear')
__global__ void fbnet_up2(const float* __restrict__ src, float* __restrict__ dst,
                          int Bn, int hi, int wi, int C) {
    int ho = hi * 2, wo = wi * 2;
    int i = blockIdx.x * 256 + threadIdx.x;
    int total = Bn * ho * wo * C;
    if (i >= total) return;
    int c = i % C; int t = i / C;
    int x = t % wo; t /= wo;
    int y = t % ho; int b = t / ho;
    float fy = (y + 0.5f) * 0.5f - 0.5f;
    float fx = (x + 0.5f) * 0.5f - 0.5f;
    float fyf = floorf(fy), fxf = floorf(fx);
    float ay = fy - fyf, ax = fx - fxf;
    int y0 = (int)fyf, x0 = (int)fxf;
    int y0c = min(max(y0, 0), hi - 1), y1c = min(max(y0 + 1, 0), hi - 1);
    int x0c = min(max(x0, 0), wi - 1), x1c = min(max(x0 + 1, 0), wi - 1);
    const float* sb = src + (size_t)b * hi * wi * C;
    float tl = sb[((size_t)y0c * wi + x0c) * C + c];
    float tr = sb[((size_t)y0c * wi + x1c) * C + c];
    float bl = sb[((size_t)y1c * wi + x0c) * C + c];
    float br = sb[((size_t)y1c * wi + x1c) * C + c];
    float top = tl + ax * (tr - tl), bot = bl + ax * (br - bl);
    dst[((size_t)(b * ho + y) * wo + x) * C + c] = top + ay * (bot - top);
}

// tfa dense_image_warp semantics (see reference _warp)
__global__ void fbnet_warp(const float* __restrict__ img, const float* __restrict__ flow,
                           float* __restrict__ dst, int Bn, int h, int w, int C,
                           int dstC, int dstOff) {
    int i = blockIdx.x * 256 + threadIdx.x;
    int total = Bn * h * w * C;
    if (i >= total) return;
    int c = i % C; int t = i / C;
    int x = t % w; t /= w;
    int y = t % h; int b = t / h;
    size_t pidx = (size_t)(b * h + y) * w + x;
    float qy = (float)y - flow[pidx * 2 + 0];
    float qx = (float)x - flow[pidx * 2 + 1];
    float fyf = fminf(fmaxf(floorf(qy), 0.f), (float)(h - 2));
    float fxf = fminf(fmaxf(floorf(qx), 0.f), (float)(w - 2));
    int fy = (int)fyf, fx = (int)fxf;
    float ay = fminf(fmaxf(qy - fyf, 0.f), 1.f);
    float ax = fminf(fmaxf(qx - fxf, 0.f), 1.f);
    const float* sb = img + ((size_t)(b * h + fy) * w + fx) * C + c;
    size_t rs = (size_t)w * C;
    float tl = sb[0], tr = sb[C], bl = sb[rs], br = sb[rs + C];
    float top = tl + ax * (tr - tl), bot = bl + ax * (br - bl);
    dst[pidx * dstC + dstOff + c] = top + ay * (bot - top);
}

__global__ void fbnet_copych(const float* __restrict__ src, float* __restrict__ dst,
                             int total, int C, int srcC, int srcOff, int dstC, int dstOff) {
    int i = blockIdx.x * 256 + threadIdx.x;
    if (i >= total) return;
    int c = i % C; size_t p = (size_t)(i / C);
    dst[p * dstC + dstOff + c] = src[p * srcC + srcOff + c];
}

__global__ void fbnet_add2(const float* __restrict__ a, const float* __restrict__ b,
                           float* __restrict__ o, int n) {
    int i = blockIdx.x * 256 + threadIdx.x;
    if (i < n) o[i] = a[i] + b[i];
}

__global__ void fbnet_add3(const float* __restrict__ a, const float* __restrict__ b,
                           const float* __restrict__ c, float* __restrict__ o, int n) {
    int i = blockIdx.x * 256 + threadIdx.x;
    if (i < n) o[i] = a[i] + b[i] + c[i];
}

// ---------------------------------------------------------------------------
// Host orchestration
// ---------------------------------------------------------------------------
extern "C" void kernel_launch(void* const* d_in, const int* in_sizes, int n_in,
                              void* d_out, int out_size, void* d_ws, size_t ws_size,
                              hipStream_t stream)
{
    (void)n_in; (void)in_sizes; (void)out_size; (void)ws_size;
    const int Bn = 4;
    static const int LH[4] = {384, 192, 96, 48};
    static const int LW[4] = {640, 320, 160, 80};
    static const int LVLC[4] = {16, 48, 112, 112};
    auto npix = [&](int l) -> size_t { return (size_t)Bn * LH[l] * LW[l]; };

    // ---- workspace carving ----
    size_t cur = 0;
    auto alloc_bytes = [&](size_t bytes) -> char* {
        cur = (cur + 255) & ~(size_t)255;
        char* p = (char*)d_ws + cur;
        cur += bytes;
        return p;
    };
    auto allocf = [&](size_t elems) -> float* { return (float*)alloc_bytes(elems * sizeof(float)); };

    // ---- f16 weight-tile region + prep ----
    _Float16* f16cur = (_Float16*)alloc_bytes((size_t)16 << 20);
    auto F = [&](int i) { return (const float*)d_in[i]; };
    auto prep = [&](int idx, int ks, int cin, int cout) -> const _Float16* {
        int Ktot = ks * ks * cin;
        int nkc = (Ktot + 63) >> 6, nnt = (cout + 15) >> 4;
        int total = nnt * nkc * 1024;
        _Float16* dst = f16cur;
        f16cur += (size_t)total;    // tile size is 2KB-multiple, alignment preserved
        fbnet_wtile<<<dim3((total + 255) / 256), dim3(256), 0, stream>>>(F(idx), dst, Ktot, cout, nkc, total);
        return dst;
    };

    // input index map (setup_inputs insertion order, depth-first flatten):
    // 0 frame1, 1 frame2, 2..7 pyr (w1,b1,w2,b2,w3,b3),
    // 8 + step*20 + dir*10 + conv*2 (+1 bias): bfe (step order = lvl 3,2,1,0; dir p12,p21; 5 convs)
    // 88.. fus: w41 b41 w42 b42 w31 b31 w32 b32 w21 b21 w22 b22 wo bo
    const _Float16 *pw1 = prep(2, 3, 3, 16), *pw2 = prep(4, 3, 16, 32), *pw3 = prep(6, 3, 32, 64);
    const float *pb1 = F(3), *pb2 = F(5), *pb3 = F(7);
    static const int lvls[4] = {3, 2, 1, 0};
    static const int fks[5]   = {3, 3, 1, 1, 1};
    static const int fco[5]   = {32, 64, 64, 16, 2};
    const _Float16* bw[4][2][5];
    const float*    bb[4][2][5];
    for (int s = 0; s < 4; ++s) {
        int cin0 = 2 * LVLC[lvls[s]];
        for (int d = 0; d < 2; ++d)
            for (int c5 = 0; c5 < 5; ++c5) {
                int base = 8 + s * 20 + d * 10 + c5 * 2;
                int cin = (c5 == 0) ? cin0 : fco[c5 - 1];
                bw[s][d][c5] = prep(base, fks[c5], cin, fco[c5]);
                bb[s][d][c5] = F(base + 1);
            }
    }
    const _Float16 *w41 = prep(88, 3, 112, 112), *w42 = prep(90, 3, 112, 112),
                   *w31 = prep(92, 3, 112, 48),  *w32 = prep(94, 3, 48, 48),
                   *w21 = prep(96, 3, 48, 16),   *w22 = prep(98, 3, 16, 16),
                   *wo_ = prep(100, 1, 16, 3);
    const float *b41 = F(89), *b42 = F(91), *b31 = F(93), *b32 = F(95),
                *b21 = F(97), *b22 = F(99), *bo = F(101);

    // ---- activation buffers ----
    float *i2f[2], *i3f[2], *i4f[2], *P[2][4], *Wrp[2][4];
    for (int f = 0; f < 2; ++f) {
        i2f[f] = allocf(npix(1) * 3);
        i3f[f] = allocf(npix(2) * 3);
        i4f[f] = allocf(npix(3) * 3);
        for (int l = 0; l < 4; ++l) {
            int C = LVLC[l];
            P[f][l]   = allocf(npix(l) * C);
            Wrp[f][l] = allocf(npix(l) * C);
        }
    }
    float* tpool = allocf(npix(1) * 32);
    float* cat   = allocf(npix(0) * 32);
    float* t1    = allocf(npix(0) * 32);
    float* t2    = allocf(npix(0) * 64);
    float* t3    = allocf(npix(0) * 64);
    float* t4    = allocf(npix(0) * 16);
    float* dflow = allocf(npix(0) * 2);
    float *fl12[4], *fl21[4];
    for (int l = 0; l < 4; ++l) { fl12[l] = allocf(npix(l) * 2); fl21[l] = allocf(npix(l) * 2); }
    float* fl12n = allocf(npix(0) * 2);
    float* fl21n = allocf(npix(0) * 2);
    float* sum   = allocf(npix(0) * 16);
    float* fA    = allocf((size_t)4 * 1024 * 1024);
    float* fB    = allocf((size_t)4 * 1024 * 1024);
    float* ubuf  = allocf(npix(0) * 16);

    // ---- launch helpers ----
    auto conv = [&](const float* in, int Cin, const _Float16* w16, const float* bias,
                    float* out, int CoutTot, int Coff, int Cout, int l, int ks, int act) {
        int h = LH[l], w = LW[l];
        int total = Bn * h * w;
        dim3 grid((total + 63) / 64, (Cout + 15) / 16);
        fbnet_conv_wmma<<<grid, dim3(128), 0, stream>>>(in, w16, bias, out, Bn, h, w,
            Cin, Cout, CoutTot, Coff, ks, (ks == 3) ? 1 : 0, act, total);
    };
    auto avgpoolL = [&](const float* src, float* dst, int lo, int C,
                        int srcC, int srcOff, int dstC, int dstOff) {
        int total = (int)(npix(lo) * C);
        fbnet_avgpool<<<dim3((total + 255) / 256), 256, 0, stream>>>(src, dst, Bn, LH[lo], LW[lo],
                                                                     C, srcC, srcOff, dstC, dstOff);
    };
    auto up2 = [&](const float* src, float* dst, int li, int C) {
        int total = (int)(npix(li) * 4 * C);
        fbnet_up2<<<dim3((total + 255) / 256), 256, 0, stream>>>(src, dst, Bn, LH[li], LW[li], C);
    };
    auto warp = [&](const float* img, const float* flow, float* dst, int l, int C,
                    int dstC, int dstOff) {
        int total = (int)(npix(l) * C);
        fbnet_warp<<<dim3((total + 255) / 256), 256, 0, stream>>>(img, flow, dst, Bn, LH[l], LW[l],
                                                                  C, dstC, dstOff);
    };
    auto copych = [&](const float* src, float* dst, int l, int C,
                      int srcC, int srcOff, int dstC, int dstOff) {
        int total = (int)(npix(l) * C);
        fbnet_copych<<<dim3((total + 255) / 256), 256, 0, stream>>>(src, dst, total, C,
                                                                    srcC, srcOff, dstC, dstOff);
    };
    auto add2 = [&](const float* a, const float* b, float* o, int n) {
        fbnet_add2<<<dim3((n + 255) / 256), 256, 0, stream>>>(a, b, o, n);
    };
    auto add3 = [&](const float* a, const float* b, const float* c, float* o, int n) {
        fbnet_add3<<<dim3((n + 255) / 256), 256, 0, stream>>>(a, b, c, o, n);
    };
    auto fill0 = [&](float* p, int n) {
        fbnet_fill0<<<dim3((n + 255) / 256), 256, 0, stream>>>(p, n);
    };
    auto flowpred = [&](const float* xin, int cin, int l,
                        const _Float16* const* Wp, const float* const* Bp, float* outflow) {
        conv(xin, cin, Wp[0], Bp[0], t1, 32, 0, 32, l, 3, ACT_RELU);
        conv(t1, 32, Wp[1], Bp[1], t2, 64, 0, 64, l, 3, ACT_RELU);
        conv(t2, 64, Wp[2], Bp[2], t3, 64, 0, 64, l, 1, ACT_RELU);
        conv(t3, 64, Wp[3], Bp[3], t4, 16, 0, 16, l, 1, ACT_RELU);
        conv(t4, 16, Wp[4], Bp[4], outflow, 2, 0, 2, l, 1, ACT_LIN);
    };

    // ---- feature pyramids (both frames) ----
    for (int f = 0; f < 2; ++f) {
        const float* fr = F(f);
        avgpoolL(fr, i2f[f], 1, 3, 3, 0, 3, 0);
        avgpoolL(i2f[f], i3f[f], 2, 3, 3, 0, 3, 0);
        avgpoolL(i3f[f], i4f[f], 3, 3, 3, 0, 3, 0);
        conv(fr,     3, pw1, pb1, P[f][0],  16, 0, 16, 0, 3, ACT_RELU);   // c11
        conv(i2f[f], 3, pw1, pb1, P[f][1],  48, 0, 16, 1, 3, ACT_RELU);   // c12
        conv(i3f[f], 3, pw1, pb1, P[f][2], 112, 0, 16, 2, 3, ACT_RELU);   // c13
        conv(i4f[f], 3, pw1, pb1, P[f][3], 112, 0, 16, 3, 3, ACT_RELU);   // c14
        avgpoolL(P[f][0], tpool, 1, 16, 16, 0, 16, 0);
        conv(tpool, 16, pw2, pb2, P[f][1],  48, 16, 32, 1, 3, ACT_RELU);  // c22
        avgpoolL(P[f][1], tpool, 2, 16, 48, 0, 16, 0);
        conv(tpool, 16, pw2, pb2, P[f][2], 112, 16, 32, 2, 3, ACT_RELU);  // c23
        avgpoolL(P[f][2], tpool, 3, 16, 112, 0, 16, 0);
        conv(tpool, 16, pw2, pb2, P[f][3], 112, 16, 32, 3, 3, ACT_RELU);  // c24
        avgpoolL(P[f][1], tpool, 2, 32, 48, 16, 32, 0);
        conv(tpool, 32, pw3, pb3, P[f][2], 112, 48, 64, 2, 3, ACT_RELU);  // c33
        avgpoolL(P[f][2], tpool, 3, 32, 112, 16, 32, 0);
        conv(tpool, 32, pw3, pb3, P[f][3], 112, 48, 64, 3, 3, ACT_RELU);  // c34
    }

    // ---- coarse-to-fine bidirectional flow estimation ----
    fill0(fl12[3], (int)(npix(3) * 2));
    fill0(fl21[3], (int)(npix(3) * 2));
    for (int s = 0; s < 4; ++s) {
        int l = lvls[s];
        int C = LVLC[l];
        int nfl = (int)(npix(l) * 2);
        const float* f1 = P[0][l];
        const float* f2 = P[1][l];
        // direction 1->2
        copych(f2, cat, l, C, C, 0, 2 * C, 0);
        warp(f1, fl12[l], cat, l, C, 2 * C, C);
        flowpred(cat, 2 * C, l, bw[s][0], bb[s][0], dflow);
        add2(fl12[l], dflow, fl12n, nfl);
        warp(f1, fl12n, Wrp[0][l], l, C, C, 0);
        // direction 2->1
        copych(f1, cat, l, C, C, 0, 2 * C, 0);
        warp(f2, fl21[l], cat, l, C, 2 * C, C);
        flowpred(cat, 2 * C, l, bw[s][1], bb[s][1], dflow);
        add2(fl21[l], dflow, fl21n, nfl);
        warp(f2, fl21n, Wrp[1][l], l, C, C, 0);
        // upsample flow for next (finer) level
        if (s < 3) {
            int nl = lvls[s + 1];
            up2(fl12n, fl12[nl], l, 2);
            up2(fl21n, fl21[nl], l, 2);
        }
    }

    // ---- fusion decoder ----
    add2(Wrp[0][3], Wrp[1][3], sum, (int)(npix(3) * 112));
    conv(sum, 112, w41, b41, fA, 112, 0, 112, 3, 3, ACT_RELU);
    conv(fA, 112, w42, b42, fB, 112, 0, 112, 3, 3, ACT_RELU);
    up2(fB, ubuf, 3, 112);
    add3(Wrp[0][2], Wrp[1][2], ubuf, sum, (int)(npix(2) * 112));
    conv(sum, 112, w31, b31, fA, 48, 0, 48, 2, 3, ACT_RELU);
    conv(fA, 48, w32, b32, fB, 48, 0, 48, 2, 3, ACT_RELU);
    up2(fB, ubuf, 2, 48);
    add3(Wrp[0][1], Wrp[1][1], ubuf, sum, (int)(npix(1) * 48));
    conv(sum, 48, w21, b21, fA, 16, 0, 16, 1, 3, ACT_RELU);
    conv(fA, 16, w22, b22, fB, 16, 0, 16, 1, 3, ACT_RELU);
    up2(fB, ubuf, 1, 16);
    add3(Wrp[0][0], Wrp[1][0], ubuf, sum, (int)(npix(0) * 16));
    conv(sum, 16, wo_, bo, (float*)d_out, 3, 0, 3, 0, 1, ACT_CLIP);
}
